// MultiQueryAttention_22823456211800
// MI455X (gfx1250) — compile-verified, matrix operand order checked by reading, not run on hardware
//
#include <hip/hip_runtime.h>
#include <hip/hip_bf16.h>
#include <stdint.h>

// ---------------------------------------------------------------------------
// Types & WMMA helpers (CDNA5 / gfx1250, wave32)
// ---------------------------------------------------------------------------
typedef __attribute__((ext_vector_type(16))) __bf16 v16bf;
typedef __attribute__((ext_vector_type(4)))  __bf16 v4bf;
typedef __attribute__((ext_vector_type(8)))  float  v8f;
typedef __attribute__((ext_vector_type(4)))  unsigned int v4u;

union Frag { v4u u[2]; v16bf v; };

// Raw workgroup barrier: only drain DScnt (LDS producer/consumer ordering),
// leaving prefetched global loads in flight across the barrier. __syncthreads
// would drain loadcnt too and destroy the software pipeline.
#define WG_BARRIER() \
  asm volatile("s_wait_dscnt 0x0\n\ts_barrier_signal -1\n\ts_barrier_wait -1" ::: "memory")

// 16-bit A (16x32) / B (32x16) fragment from a row-major [16][>=32] tile:
// lane supplies row (lane&15); per-lane element i maps to
// k = (lane>>4)*8 + i  (i<8)   and   k = 16 + (lane>>4)*8 + (i-8)  (i>=8)
__device__ __forceinline__ v16bf load_frag_row(const __bf16* rowp, int half) {
  Frag f;
  f.u[0] = *(const v4u*)(rowp + half * 8);
  f.u[1] = *(const v4u*)(rowp + 16 + half * 8);
  return f.v;
}

// Prepacked fragment: 32 lanes x 16 bf16 contiguous per lane (1KB block).
__device__ __forceinline__ v16bf load_frag_packed(const __bf16* p) {
  Frag f;
  f.u[0] = *(const v4u*)(p);
  f.u[1] = *(const v4u*)(p + 8);
  return f.v;
}

__device__ __forceinline__ v8f wmma_bf16(v16bf a, v16bf b, v8f c) {
  // D = A(16x32) * B(32x16) + C, f32 accumulate
  return __builtin_amdgcn_wmma_f32_16x16x32_bf16(false, a, false, b, (short)0, c,
                                                 false, false);
}

// ---------------------------------------------------------------------------
// Weight prepack: fp32 [O][Kd] -> bf16 fragment-ordered blocks
// block (kt, nt) -> 512 bf16 at ((kt*nnt + nt)*512); lane-contiguous.
// ---------------------------------------------------------------------------
__global__ void pack_w_kernel(const float* __restrict__ src, __bf16* __restrict__ dst,
                              int Kd, int nnt) {
  const int bid = blockIdx.x;
  const int nkt = Kd >> 5;
  const int kt = bid % nkt;
  const int nt = bid / nkt;
  const int lane = threadIdx.x;  // 32 threads
  const int o = nt * 16 + (lane & 15);
  const int h = lane >> 4;
  __bf16* d = dst + ((size_t)(kt * nnt + nt)) * 512 + lane * 16;
  const float* srow = src + (size_t)o * Kd + kt * 32;
#pragma unroll
  for (int i = 0; i < 16; ++i) {
    const int k = (i < 8) ? (h * 8 + i) : (16 + h * 8 + (i - 8));
    d[i] = (__bf16)srow[k];
  }
}

// ---------------------------------------------------------------------------
// Patch-embed conv as implicit GEMM:
//   out[t][o] = sum_k patch(t)[k] * w[o][k],  k = c*64 + dz*16 + hy*4 + wx
// Block: 256 thr (8 waves); M=32 tokens/WG, N=384 (wave -> 3 n-tiles), K=24576.
// Manually 2x-unrolled K loop: static dual LDS buffers + static dual B-frag
// register sets (no rotation copies), raw DScnt-only barriers so A/B global
// prefetches stay in flight across the barrier.
// ---------------------------------------------------------------------------
__global__ __launch_bounds__(256) void conv_gemm_kernel(
    const float* __restrict__ x, const __bf16* __restrict__ wp,
    const float* __restrict__ bias, float* __restrict__ out) {
  constexpr int ROW = 40;               // padded LDS row stride (bf16 units)
  __shared__ __bf16 As0[32 * ROW];
  __shared__ __bf16 As1[32 * ROW];
  const int tid = threadIdx.x;
  const int wave = tid >> 5, lane = tid & 31;
  const int lrow = lane & 15, lh = lane >> 4;
  const int t0 = blockIdx.x * 32;
  const int nt0 = wave * 3;
  v8f acc[2][3] = {};

  // A loader: lane -> token (tw-adjacent => coalesced), wave -> patch row
  const int m_ld = lane;                // token within tile
  const int r_ld = wave;                // patch row 0..7
  const int tok = t0 + m_ld;
  const int td = tok >> 8, th = (tok >> 4) & 15, tw = tok & 15;
  const int hy = r_ld & 3;

  auto a_ptr = [&](int kt) {
    const int c = kt >> 1, half = kt & 1;
    const int dz = half * 2 + (r_ld >> 2);
    return (const float4*)(x + (((size_t)(c * 64 + td * 4 + dz) * 64 +
                                 (th * 4 + hy)) * 64 + tw * 4));
  };
  auto b_ptr = [&](int kt) {
    return wp + ((size_t)(kt * 24 + nt0)) * 512 + lane * 16;
  };
  auto store_a = [&](__bf16* buf, const float4& v) {
    v4bf hb;
    hb[0] = (__bf16)v.x; hb[1] = (__bf16)v.y;
    hb[2] = (__bf16)v.z; hb[3] = (__bf16)v.w;
    *(v4bf*)(&buf[m_ld * ROW + r_ld * 4]) = hb;
  };
  auto compute = [&](const __bf16* buf, v16bf B0, v16bf B1, v16bf B2) {
#pragma unroll
    for (int ms = 0; ms < 2; ++ms) {
      const v16bf a = load_frag_row(&buf[(ms * 16 + lrow) * ROW], lh);
      acc[ms][0] = wmma_bf16(a, B0, acc[ms][0]);
      acc[ms][1] = wmma_bf16(a, B1, acc[ms][1]);
      acc[ms][2] = wmma_bf16(a, B2, acc[ms][2]);
    }
  };

  float4 areg = *a_ptr(0);
  v16bf bE0, bE1, bE2, bO0, bO1, bO2;
  {
    const __bf16* bb = b_ptr(0);
    bE0 = load_frag_packed(bb);
    bE1 = load_frag_packed(bb + 512);
    bE2 = load_frag_packed(bb + 1024);
  }

  for (int kt = 0; kt < 768; kt += 2) {
    // ---- even step: consume As0 + bE*, prefetch A(kt+1) and bO* = B(kt+1)
    store_a(As0, areg);
    areg = *a_ptr(kt + 1);
    WG_BARRIER();
    {
      const __bf16* bb = b_ptr(kt + 1);
      bO0 = load_frag_packed(bb);
      bO1 = load_frag_packed(bb + 512);
      bO2 = load_frag_packed(bb + 1024);
    }
    compute(As0, bE0, bE1, bE2);

    // ---- odd step: consume As1 + bO*, prefetch A/B for kt+2 (clamped tail)
    store_a(As1, areg);
    const int ktn = (kt + 2 < 768) ? (kt + 2) : 767;
    areg = *a_ptr(ktn);
    WG_BARRIER();
    {
      const __bf16* bb = b_ptr(ktn);
      bE0 = load_frag_packed(bb);
      bE1 = load_frag_packed(bb + 512);
      bE2 = load_frag_packed(bb + 1024);
    }
    compute(As1, bO0, bO1, bO2);
  }

#pragma unroll
  for (int ms = 0; ms < 2; ++ms)
#pragma unroll
    for (int j = 0; j < 3; ++j) {
      const int n = (nt0 + j) * 16 + lrow;
      const float bs = bias[n];
#pragma unroll
      for (int i = 0; i < 8; ++i) {
        const int t = t0 + ms * 16 + i + lh * 8;
        out[(size_t)t * 384 + n] = acc[ms][j][i] + bs;
      }
    }
}

// ---------------------------------------------------------------------------
// LayerNorm over C=384 per token; fp32 in -> bf16 out. 128 threads/token.
// ---------------------------------------------------------------------------
__global__ __launch_bounds__(128) void ln_kernel(const float* __restrict__ x,
                                                 const float* __restrict__ g,
                                                 const float* __restrict__ b,
                                                 __bf16* __restrict__ y) {
  __shared__ float red[128];
  const int t = blockIdx.x, tid = threadIdx.x;
  const float* xr = x + (size_t)t * 384;
  const float v0 = xr[tid], v1 = xr[tid + 128], v2 = xr[tid + 256];
  red[tid] = v0 + v1 + v2;
  __syncthreads();
  for (int o = 64; o > 0; o >>= 1) { if (tid < o) red[tid] += red[tid + o]; __syncthreads(); }
  const float mean = red[0] * (1.f / 384.f);
  __syncthreads();
  const float d0 = v0 - mean, d1 = v1 - mean, d2 = v2 - mean;
  red[tid] = d0 * d0 + d1 * d1 + d2 * d2;
  __syncthreads();
  for (int o = 64; o > 0; o >>= 1) { if (tid < o) red[tid] += red[tid + o]; __syncthreads(); }
  const float rstd = rsqrtf(red[0] * (1.f / 384.f) + 1e-5f);
  __bf16* yr = y + (size_t)t * 384;
  yr[tid]       = (__bf16)(d0 * rstd * g[tid]       + b[tid]);
  yr[tid + 128] = (__bf16)(d1 * rstd * g[tid + 128] + b[tid + 128]);
  yr[tid + 256] = (__bf16)(d2 * rstd * g[tid + 256] + b[tid + 256]);
}

__global__ void zero_kernel(unsigned int* __restrict__ p, long n) {
  const long i = (long)blockIdx.x * 256 + threadIdx.x;
  if (i < n) p[i] = 0u;
}

// ---------------------------------------------------------------------------
// Generic projection GEMM: A bf16 [4096][384] x packed W -> out.
// mode 0: bf16 out[t*ostride + n]
// mode 1: bf16 out[n*ostride + t]            (transposed, for V^T)
// mode 2: bf16 out[t*ostride + (n/48)*64+n%48] (head-padded Q)
// mode 3: f32  out[n*ostride + t]            (O-proj, channel-major)
// ---------------------------------------------------------------------------
__global__ __launch_bounds__(256) void gemm_proj_kernel(
    const __bf16* __restrict__ A, const __bf16* __restrict__ wp,
    const float* __restrict__ bias, void* __restrict__ outp,
    int Ntiles, int ostride, int mode) {
  __shared__ __bf16 As[32 * 32];
  const int tid = threadIdx.x;
  const int wave = tid >> 5, lane = tid & 31;
  const int lrow = lane & 15, lh = lane >> 4;
  const int t0 = blockIdx.x * 32;
  const int ntb = (wave * Ntiles) >> 3;
  const int ncnt = (((wave + 1) * Ntiles) >> 3) - ntb;  // 0..3
  v8f acc[2][3] = {};
  for (int kt = 0; kt < 12; ++kt) {
    __syncthreads();
    if (tid < 128) {
      const int row = tid >> 2, ch = tid & 3;
      *(v4u*)(&As[row * 32 + ch * 8]) =
          *(const v4u*)(A + (size_t)(t0 + row) * 384 + kt * 32 + ch * 8);
    }
    __syncthreads();
    v16bf bfr[3] = {};
#pragma unroll
    for (int j = 0; j < 3; ++j)
      if (j < ncnt)
        bfr[j] = load_frag_packed(wp + ((size_t)(kt * Ntiles + ntb + j)) * 512 + lane * 16);
#pragma unroll
    for (int ms = 0; ms < 2; ++ms) {
      const v16bf a = load_frag_row(&As[(ms * 16 + lrow) * 32], lh);
#pragma unroll
      for (int j = 0; j < 3; ++j)
        if (j < ncnt) acc[ms][j] = wmma_bf16(a, bfr[j], acc[ms][j]);
    }
  }
#pragma unroll
  for (int ms = 0; ms < 2; ++ms)
#pragma unroll
    for (int j = 0; j < 3; ++j) {
      if (j >= ncnt) continue;
      const int n = (ntb + j) * 16 + lrow;
      const float bs = bias[n];
#pragma unroll
      for (int i = 0; i < 8; ++i) {
        const int t = t0 + ms * 16 + i + lh * 8;
        const float val = acc[ms][j][i] + bs;
        if (mode == 0)      ((__bf16*)outp)[(size_t)t * ostride + n] = (__bf16)val;
        else if (mode == 1) ((__bf16*)outp)[(size_t)n * ostride + t] = (__bf16)val;
        else if (mode == 2) ((__bf16*)outp)[(size_t)t * ostride + (n / 48) * 64 + (n % 48)] = (__bf16)val;
        else                ((float*)outp)[(size_t)n * ostride + t] = val;
      }
    }
}

// ---------------------------------------------------------------------------
// Flash MQA attention. One wave per (head, 16-query tile); block = 8 waves.
// Scores computed transposed (C = [s][t]) so softmax stats are lane-local and
// the C-layout P^T converts to the PV B-fragment with zero cross-lane traffic.
// V frags issued + next K tile prefetched before the softmax VALU so the
// global loads overlap the exp() work.
//   Qp: [4096][512] bf16 (head h at cols h*64, d 48..63 zero-padded)
//   Kp: [4096][64]  bf16 (d 48..63 zero)
//   Vt: [48][4096]  bf16 (transposed V)
//   out: [4096][384] bf16 (head h at cols h*48)
// ---------------------------------------------------------------------------
__global__ __launch_bounds__(256) void attn_kernel(
    const __bf16* __restrict__ Qp, const __bf16* __restrict__ Kp,
    const __bf16* __restrict__ Vt, __bf16* __restrict__ outp) {
  const int tid = threadIdx.x;
  const int wave = tid >> 5, lane = tid & 31;
  const int lrow = lane & 15, lh = lane >> 4;
  const int h = blockIdx.x >> 5;
  const int t0 = (blockIdx.x & 31) * 128 + wave * 16;

  const __bf16* qrow = Qp + (size_t)(t0 + lrow) * 512 + h * 64;
  const v16bf qb0 = load_frag_row(qrow, lh);       // d = 0..31
  const v16bf qb1 = load_frag_row(qrow + 32, lh);  // d = 32..63 (48..63 zero in K)

  v8f o0 = {}, o1 = {}, o2 = {};
  float mst = -3.0e38f, lsum = 0.f;
  const float sc = 0.14433756729740643f;  // 1/sqrt(48)

  const __bf16* k_base = Kp + (size_t)lrow * 64;
  const __bf16* v_base = Vt + (size_t)lrow * 4096;
  auto kfrag = [&](int s, int sub, int dchunk) {
    return load_frag_row(k_base + (size_t)(s + sub * 16) * 64 + dchunk * 32, lh);
  };
  v16bf ka0 = kfrag(0, 0, 0), ka1 = kfrag(0, 0, 1);
  v16bf kb0 = kfrag(0, 1, 0), kb1 = kfrag(0, 1, 1);

  for (int s0 = 0; s0 < 4096; s0 += 32) {
    v8f c0 = {}, c1 = {};
    c0 = wmma_bf16(ka0, qb0, c0);
    c0 = wmma_bf16(ka1, qb1, c0);
    c1 = wmma_bf16(kb0, qb0, c1);
    c1 = wmma_bf16(kb1, qb1, c1);

    // current-tile V frags + next-tile K frags, issued before softmax VALU
    const v16bf vf0 = load_frag_row(v_base + s0, lh);
    const v16bf vf1 = load_frag_row(v_base + 16 * 4096 + s0, lh);
    const v16bf vf2 = load_frag_row(v_base + 32 * 4096 + s0, lh);
    const int sn = (s0 + 32 < 4096) ? (s0 + 32) : 0;
    ka0 = kfrag(sn, 0, 0); ka1 = kfrag(sn, 0, 1);
    kb0 = kfrag(sn, 1, 0); kb1 = kfrag(sn, 1, 1);

    float tmax = -3.0e38f;
    float s0v[8], s1v[8];
#pragma unroll
    for (int i = 0; i < 8; ++i) {
      s0v[i] = c0[i] * sc;
      s1v[i] = c1[i] * sc;
      tmax = fmaxf(tmax, fmaxf(s0v[i], s1v[i]));
    }
    tmax = fmaxf(tmax, __shfl_xor(tmax, 16));
    const float mnew = fmaxf(mst, tmax);
    const float corr = __expf(mst - mnew);
    float e0[8], e1[8], ps = 0.f;
#pragma unroll
    for (int i = 0; i < 8; ++i) {
      e0[i] = __expf(s0v[i] - mnew);
      e1[i] = __expf(s1v[i] - mnew);
      ps += e0[i] + e1[i];
    }
    ps += __shfl_xor(ps, 16);
    lsum = lsum * corr + ps;
    mst = mnew;

    v16bf pb;
#pragma unroll
    for (int i = 0; i < 8; ++i) {
      pb[i]     = (__bf16)e0[i];
      pb[i + 8] = (__bf16)e1[i];
      o0[i] *= corr; o1[i] *= corr; o2[i] *= corr;
    }
    o0 = wmma_bf16(vf0, pb, o0);
    o1 = wmma_bf16(vf1, pb, o1);
    o2 = wmma_bf16(vf2, pb, o2);
  }
  const float inv = 1.0f / lsum;
  __bf16* orow = outp + (size_t)(t0 + lrow) * 384 + h * 48;
#pragma unroll
  for (int i = 0; i < 8; ++i) {
    orow[0 * 16 + i + lh * 8] = (__bf16)(o0[i] * inv);
    orow[1 * 16 + i + lh * 8] = (__bf16)(o1[i] * inv);
    orow[2 * 16 + i + lh * 8] = (__bf16)(o2[i] * inv);
  }
}

// ---------------------------------------------------------------------------
// Trilinear upsample (align_corners=True, 16->64 per axis) + residual add.
// 4 outputs per thread (float4 residual read / float4 output write).
// ---------------------------------------------------------------------------
__global__ __launch_bounds__(256) void upsample_kernel(
    const float* __restrict__ s, const float* __restrict__ res,
    float* __restrict__ out) {
  const long q = (long)blockIdx.x * 256 + threadIdx.x;
  const long idx = q * 4;
  const int xb = (int)(idx & 63);
  const int y = (int)((idx >> 6) & 63), z = (int)((idx >> 12) & 63);
  const int c = (int)(idx >> 18);
  const float f = 15.f / 63.f;
  const float fy = y * f, fz = z * f;
  const int y0 = (int)fy, z0 = (int)fz;
  const float wy = fy - y0, wz = fz - z0;
  const int y1 = min(y0 + 1, 15), z1 = min(z0 + 1, 15);
  const float* bp = s + (size_t)c * 4096;
  const float* p00 = bp + z0 * 256 + y0 * 16;
  const float* p01 = bp + z0 * 256 + y1 * 16;
  const float* p10 = bp + z1 * 256 + y0 * 16;
  const float* p11 = bp + z1 * 256 + y1 * 16;
  const float w00 = (1 - wz) * (1 - wy), w01 = (1 - wz) * wy;
  const float w10 = wz * (1 - wy),       w11 = wz * wy;
  const float4 r = *(const float4*)(res + idx);
  float o[4];
#pragma unroll
  for (int j = 0; j < 4; ++j) {
    const float fx = (xb + j) * f;
    const int x0 = (int)fx;
    const float wx = fx - x0;
    const int x1 = min(x0 + 1, 15);
    const float a = p00[x0] * w00 + p01[x0] * w01 + p10[x0] * w10 + p11[x0] * w11;
    const float b = p00[x1] * w00 + p01[x1] * w01 + p10[x1] * w10 + p11[x1] * w11;
    o[j] = a * (1 - wx) + b * wx;
  }
  float4 ov;
  ov.x = o[0] + r.x; ov.y = o[1] + r.y; ov.z = o[2] + r.z; ov.w = o[3] + r.w;
  *(float4*)(out + idx) = ov;
}

// ---------------------------------------------------------------------------
// Host orchestration
// ---------------------------------------------------------------------------
extern "C" void kernel_launch(void* const* d_in, const int* in_sizes, int n_in,
                              void* d_out, int out_size, void* d_ws, size_t ws_size,
                              hipStream_t stream) {
  const float* tensor1  = (const float*)d_in[0];
  const float* tensor2  = (const float*)d_in[1];
  const float* w_patch1 = (const float*)d_in[2];
  const float* b_patch1 = (const float*)d_in[3];
  const float* w_patch2 = (const float*)d_in[4];
  const float* b_patch2 = (const float*)d_in[5];
  const float* ln1_g = (const float*)d_in[6];
  const float* ln1_b = (const float*)d_in[7];
  const float* ln2_g = (const float*)d_in[8];
  const float* ln2_b = (const float*)d_in[9];
  const float* wq = (const float*)d_in[10];
  const float* bq = (const float*)d_in[11];
  const float* wk = (const float*)d_in[12];
  const float* bk = (const float*)d_in[13];
  const float* wv = (const float*)d_in[14];
  const float* bv = (const float*)d_in[15];
  const float* wo = (const float*)d_in[16];
  const float* bo = (const float*)d_in[17];

  char* ws = (char*)d_ws;
  size_t off = 0;
  auto alloc = [&](size_t bytes) {
    char* p = ws + off;
    off += (bytes + 255) & ~(size_t)255;
    return p;
  };
  __bf16* wb1 = (__bf16*)alloc((size_t)768 * 24 * 512 * 2);
  __bf16* wb2 = (__bf16*)alloc((size_t)768 * 24 * 512 * 2);
  __bf16* wqp = (__bf16*)alloc((size_t)12 * 24 * 512 * 2);
  __bf16* wkp = (__bf16*)alloc((size_t)12 * 3 * 512 * 2);
  __bf16* wvp = (__bf16*)alloc((size_t)12 * 3 * 512 * 2);
  __bf16* wop = (__bf16*)alloc((size_t)12 * 24 * 512 * 2);
  float*  t1r = (float*)alloc((size_t)4096 * 384 * 4);
  float*  t2r = (float*)alloc((size_t)4096 * 384 * 4);
  __bf16* x1  = (__bf16*)alloc((size_t)4096 * 384 * 2);
  __bf16* x2  = (__bf16*)alloc((size_t)4096 * 384 * 2);
  __bf16* Qp  = (__bf16*)alloc((size_t)4096 * 512 * 2);
  __bf16* Kp  = (__bf16*)alloc((size_t)4096 * 64 * 2);
  __bf16* Vt  = (__bf16*)alloc((size_t)48 * 4096 * 2);
  __bf16* attn = (__bf16*)alloc((size_t)4096 * 384 * 2);
  float*  proj = (float*)alloc((size_t)384 * 4096 * 4);

  // weight prepack (bf16, fragment order)
  pack_w_kernel<<<768 * 24, 32, 0, stream>>>(w_patch1, wb1, 24576, 24);
  pack_w_kernel<<<768 * 24, 32, 0, stream>>>(w_patch2, wb2, 24576, 24);
  pack_w_kernel<<<12 * 24, 32, 0, stream>>>(wq, wqp, 384, 24);
  pack_w_kernel<<<12 * 3, 32, 0, stream>>>(wk, wkp, 384, 3);
  pack_w_kernel<<<12 * 3, 32, 0, stream>>>(wv, wvp, 384, 3);
  pack_w_kernel<<<12 * 24, 32, 0, stream>>>(wo, wop, 384, 24);

  // patch-embed convs (implicit GEMM, WMMA bf16)
  conv_gemm_kernel<<<128, 256, 0, stream>>>(tensor1, wb1, b_patch1, t1r);
  conv_gemm_kernel<<<128, 256, 0, stream>>>(tensor2, wb2, b_patch2, t2r);
  ln_kernel<<<4096, 128, 0, stream>>>(t1r, ln1_g, ln1_b, x1);
  ln_kernel<<<4096, 128, 0, stream>>>(t2r, ln2_g, ln2_b, x2);

  // zero head-pad regions of Qp and Kp
  zero_kernel<<<(1048576 + 255) / 256, 256, 0, stream>>>((unsigned int*)Qp, 1048576);
  zero_kernel<<<(131072 + 255) / 256, 256, 0, stream>>>((unsigned int*)Kp, 131072);

  // projections
  gemm_proj_kernel<<<128, 256, 0, stream>>>(x1, wqp, bq, Qp, 24, 512, 2);   // Q head-padded
  gemm_proj_kernel<<<128, 256, 0, stream>>>(x2, wkp, bk, Kp, 3, 64, 0);     // K padded rows
  gemm_proj_kernel<<<128, 256, 0, stream>>>(x2, wvp, bv, Vt, 3, 4096, 1);   // V transposed

  // flash MQA attention
  attn_kernel<<<256, 256, 0, stream>>>(Qp, Kp, Vt, attn);

  // output projection, written channel-major fp32 [384][16^3]
  gemm_proj_kernel<<<128, 256, 0, stream>>>(attn, wop, bo, proj, 24, 4096, 3);

  // trilinear upsample + residual -> d_out fp32 [1,384,64,64,64]
  upsample_kernel<<<98304, 256, 0, stream>>>(proj, tensor2, (float*)d_out);
}
